// BahdanauAttention_30391188587358
// MI455X (gfx1250) — matrix operand order repaired and checked by reading, not compiled, not run on hardware
//
#include <hip/hip_runtime.h>
#include <hip/hip_bf16.h>

#define Bc   32
#define Lc   2048
#define ENCc 512
#define DECc 512

typedef __attribute__((ext_vector_type(16))) __bf16 v16bf;
typedef __attribute__((ext_vector_type(8)))  __bf16 v8bf;
typedef __attribute__((ext_vector_type(4)))  __bf16 v4bf;
typedef __attribute__((ext_vector_type(8)))  float  v8f;

// A-fragment: lane<16 needs K kk+0..7 and kk+16..23 of its row (two 16B LDS reads)
__device__ __forceinline__ v16bf ld_split(const __bf16* p, int lo) {
  v8bf a = *(const v8bf*)(p + lo);
  v8bf b = *(const v8bf*)(p + lo + 16);
  v16bf r;
#pragma unroll
  for (int i = 0; i < 8; ++i) { r[i] = a[i]; r[i + 8] = b[i]; }
  return r;
}

// 16 consecutive K from p (unused; B fragments use ld_split like A)
__device__ __forceinline__ v16bf ld_contig(const __bf16* p) {
  v8bf a = *(const v8bf*)(p);
  v8bf b = *(const v8bf*)(p + 8);
  v16bf r;
#pragma unroll
  for (int i = 0; i < 8; ++i) { r[i] = a[i]; r[i + 8] = b[i]; }
  return r;
}

// ---------------- dec_features = dec_state @ W_s.T + b_s  [B, DEC] ----------------
__global__ __launch_bounds__(256)
void k_decfeat(const float* __restrict__ dec_state, const float* __restrict__ W_s,
               const float* __restrict__ b_s, float* __restrict__ decf) {
  const int idx = blockIdx.x * 256 + threadIdx.x;   // B*DEC total
  const int b = idx >> 9;
  const int d = idx & 511;
  const float4* xs = (const float4*)(dec_state + (size_t)b * DECc);
  const float4* ws = (const float4*)(W_s + (size_t)d * DECc);
  float acc = b_s[d];
#pragma unroll 4
  for (int k = 0; k < DECc / 4; ++k) {
    float4 x = xs[k], w = ws[k];
    acc += x.x * w.x + x.y * w.y + x.z * w.z + x.w * w.w;
  }
  decf[idx] = acc;
}

// ---------------- scores[b,l] via bf16 WMMA GEMM + fused tanh/score ----------------
#define ROWS 64
#define AS   520   // padded bf16 row stride (bank-conflict-free)
#define BSR  520

__global__ __launch_bounds__(128)
void k_scores(const float* __restrict__ enc, const float* __restrict__ W_h,
              const float* __restrict__ decf, const float* __restrict__ cov,
              const float* __restrict__ W_c, const float* __restrict__ vvec,
              float* __restrict__ scores) {
  __shared__ __bf16 Atile[ROWS * AS];   // 66,560 B
  __shared__ __bf16 Btile[16 * BSR];    // 16,640 B

  const int bx  = blockIdx.x;
  const int b   = bx >> 5;              // 32 l-tiles per batch
  const int l0  = (bx & 31) * ROWS;
  const int tid = threadIdx.x;
  const int wv   = tid >> 5;
  const int lane = tid & 31;
  const int m    = lane & 15;           // A row in tile / B column (d)
  const int hs   = lane >> 4;           // lane-half select

  // ---- stage A tile: enc[b, l0:l0+64, :] fp32 -> bf16 LDS ----
  const float* encB = enc + ((size_t)b * Lc + l0) * ENCc;
#pragma unroll 4
  for (int i = tid; i < ROWS * ENCc / 4; i += 128) {
    const int idx4 = i * 4;
    const int row = idx4 >> 9;
    const int col = idx4 & 511;
    const float4 f = *(const float4*)(encB + idx4);
    v4bf h; h[0] = (__bf16)f.x; h[1] = (__bf16)f.y; h[2] = (__bf16)f.z; h[3] = (__bf16)f.w;
    *(v4bf*)(&Atile[row * AS + col]) = h;
  }

  const int wrow0 = wv * 16;
  const int rbase = l0 + wrow0 + hs * 8;   // C layout: lane-half selects M+8
  float covv[8];
#pragma unroll
  for (int r = 0; r < 8; ++r) covv[r] = cov[b * Lc + rbase + r];

  float sacc[8];
#pragma unroll
  for (int r = 0; r < 8; ++r) sacc[r] = 0.0f;

  for (int dt = 0; dt < DECc / 16; ++dt) {
    __syncthreads();
    // ---- stage B tile: W_h[dt*16 .. dt*16+15, :] fp32 -> bf16 LDS ----
    const float* WhT = W_h + (size_t)(dt * 16) * ENCc;
#pragma unroll 4
    for (int i = tid; i < 16 * ENCc / 4; i += 128) {
      const int idx4 = i * 4;
      const int row = idx4 >> 9;
      const int col = idx4 & 511;
      const float4 f = *(const float4*)(WhT + idx4);
      v4bf h; h[0] = (__bf16)f.x; h[1] = (__bf16)f.y; h[2] = (__bf16)f.z; h[3] = (__bf16)f.w;
      *(v4bf*)(&Btile[row * BSR + col]) = h;
    }
    __syncthreads();

    const __bf16* arow = &Atile[(wrow0 + m) * AS];
    const __bf16* brow = &Btile[m * BSR];
    v8f c = {};
#pragma unroll
    for (int kk = 0; kk < ENCc / 32; ++kk) {
      v16bf af = ld_split(arow, kk * 32 + hs * 8);
      v16bf bf = ld_split(brow, kk * 32 + hs * 8);
      c = __builtin_amdgcn_wmma_f32_16x16x32_bf16(false, af, false, bf,
                                                  (short)0, c, false, false);
    }

    // fuse: tanh(encf + decf[d] + cov*W_c[d]) * v[d], accumulated per row
    const int d = dt * 16 + m;
    const float dfd = decf[b * DECc + d];
    const float wc  = W_c[d];
    const float vd  = vvec[d];
#pragma unroll
    for (int r = 0; r < 8; ++r) {
      const float t = tanhf(c[r] + dfd + covv[r] * wc);
      sacc[r] += t * vd;
    }
  }

  // fold the 16 d-lanes of each half-wave into per-row score sums
#pragma unroll
  for (int r = 0; r < 8; ++r) {
    float a = sacc[r];
    a += __shfl_xor(a, 1, 32);
    a += __shfl_xor(a, 2, 32);
    a += __shfl_xor(a, 4, 32);
    a += __shfl_xor(a, 8, 32);
    sacc[r] = a;
  }
  if (m == 0) {
#pragma unroll
    for (int r = 0; r < 8; ++r) scores[b * Lc + rbase + r] = sacc[r];
  }
}

// ---------------- masked softmax over L per batch ----------------
__global__ __launch_bounds__(256)
void k_softmax(const float* __restrict__ scores, const int* __restrict__ mask,
               float* __restrict__ attn) {
  const int b = blockIdx.x;
  const int tid = threadIdx.x;
  __shared__ float red[256];
  float vals[8];
  float mx = -3.0e38f;
#pragma unroll
  for (int i = 0; i < 8; ++i) {
    const int l = tid + 256 * i;
    float s = scores[b * Lc + l];
    s = (mask[b * Lc + l] == 0) ? -1.0e9f : s;
    vals[i] = s;
    mx = fmaxf(mx, s);
  }
  red[tid] = mx; __syncthreads();
  for (int s = 128; s > 0; s >>= 1) {
    if (tid < s) red[tid] = fmaxf(red[tid], red[tid + s]);
    __syncthreads();
  }
  mx = red[0]; __syncthreads();
  float sum = 0.f;
#pragma unroll
  for (int i = 0; i < 8; ++i) { vals[i] = __expf(vals[i] - mx); sum += vals[i]; }
  red[tid] = sum; __syncthreads();
  for (int s = 128; s > 0; s >>= 1) {
    if (tid < s) red[tid] += red[tid + s];
    __syncthreads();
  }
  const float inv = 1.0f / red[0];
#pragma unroll
  for (int i = 0; i < 8; ++i) attn[b * Lc + tid + 256 * i] = vals[i] * inv;
}

// ---------------- context[b,e] = sum_l attn[b,l] * enc[b,l,e] ----------------
__global__ __launch_bounds__(256)
void k_context(const float* __restrict__ enc, const float* __restrict__ attn,
               float* __restrict__ ctx) {
  const int b  = blockIdx.x;
  const int ec = blockIdx.y;                 // 8 chunks of 64 e-columns
  const int tid = threadIdx.x;
  const int e = ec * 64 + (tid & 63);
  const int g = tid >> 6;                    // 4 l-interleave groups
  const float* encB = enc + (size_t)b * Lc * ENCc;
  const float* ab   = attn + b * Lc;
  float acc = 0.f;
#pragma unroll 4
  for (int l = g; l < Lc; l += 4) acc += ab[l] * encB[(size_t)l * ENCc + e];
  __shared__ float red[256];
  red[tid] = acc; __syncthreads();
  if (g == 0) ctx[b * ENCc + e] = red[tid] + red[tid + 64] + red[tid + 128] + red[tid + 192];
}

extern "C" void kernel_launch(void* const* d_in, const int* in_sizes, int n_in,
                              void* d_out, int out_size, void* d_ws, size_t ws_size,
                              hipStream_t stream) {
  (void)in_sizes; (void)n_in; (void)out_size; (void)ws_size;
  const float* enc       = (const float*)d_in[0];  // [B,L,ENC]
  const float* dec_state = (const float*)d_in[1];  // [B,DEC]
  const float* coverage  = (const float*)d_in[2];  // [B,L]
  const int*   mask      = (const int*)  d_in[3];  // [B,L]
  const float* W_h       = (const float*)d_in[4];  // [DEC,ENC]
  const float* W_s       = (const float*)d_in[5];  // [DEC,DEC]
  const float* b_s       = (const float*)d_in[6];  // [DEC]
  const float* W_c       = (const float*)d_in[7];  // [DEC,1]
  const float* vvec      = (const float*)d_in[8];  // [1,DEC]

  float* out  = (float*)d_out;
  float* ctx  = out;                 // [B,ENC]
  float* attn = out + Bc * ENCc;     // [B,L]

  float* decf   = (float*)d_ws;          // [B,DEC]  = 64 KB
  float* scores = decf + Bc * DECc;      // [B,L]    = 256 KB

  k_decfeat<<<(Bc * DECc) / 256, 256, 0, stream>>>(dec_state, W_s, b_s, decf);
  k_scores<<<Bc * (Lc / ROWS), 128, 0, stream>>>(enc, W_h, decf, coverage, W_c, vvec, scores);
  k_softmax<<<Bc, 256, 0, stream>>>(scores, mask, attn);
  k_context<<<dim3(Bc, ENCc / 64), 256, 0, stream>>>(enc, attn, ctx);
}
